// CosineAttention_40037685133428
// MI455X (gfx1250) — compile-verified, matrix operand order checked by reading, not run on hardware
//
#include <hip/hip_runtime.h>
#include <hip/hip_bf16.h>

typedef __attribute__((ext_vector_type(16))) __bf16 v16bf;
typedef __attribute__((ext_vector_type(8)))  float  v8f;

#define HEADS 8
#define DH    64
#define DIM   512
#define SEQ   2048
#define BATCH 2
#define ROWS  (BATCH * SEQ)   // 4096
#define QKSCALE 0.125f        // 64^-0.5
#define EPS 1e-4f

#if __has_builtin(__builtin_amdgcn_s_wait_asynccnt)
#define ASYNC_WAIT(n) __builtin_amdgcn_s_wait_asynccnt(n)
#else
#define ASYNC_WAIT(n) asm volatile("s_wait_asynccnt %0" :: "i"(n) : "memory")
#endif

__device__ __forceinline__ v8f wmma_bf16(v16bf a, v16bf b, v8f c) {
  // (neg_a, A, neg_b, B, c_mod, C, reuse_a, reuse_b)
  return __builtin_amdgcn_wmma_f32_16x16x32_bf16(false, a, false, b, (short)0, c, false, false);
}

// Async DMA: copy 16 bytes global -> LDS per lane (GLOBAL_LOAD_ASYNC_TO_LDS_B128,
// tracked with ASYNCcnt). lds_off = workgroup-relative LDS byte address (low 32
// bits of the generic pointer to a __shared__ object).
__device__ __forceinline__ void async_copy16(uint32_t lds_off, const __bf16* g) {
  uint64_t ga = (uint64_t)(uintptr_t)g;
  asm volatile("global_load_async_to_lds_b128 %0, %1, off"
               :: "v"(lds_off), "v"(ga) : "memory");
}
__device__ __forceinline__ uint32_t lds_addr(const void* p) {
  return (uint32_t)(uintptr_t)p;
}

// Load one 16x32 bf16 operand tile in the CDNA5 A/B register layout (v16bf).
// A-matrix: rc0 = first matrix row; lane L holds row (rc0 + L%16), K-chunks
//   (L/16)*8 + {0..7} and 16 + (L/16)*8 + {0..7}.
// B-matrix (act @ W^T): rc0 = first B column == a row of the row-major [N,K]
//   operand, same addressing. Both 8-element groups contiguous -> two 16B loads.
__device__ __forceinline__ v16bf load_tile16(const __bf16* base, int rc0, int ld,
                                             int k0, int lane) {
  const __bf16* p = base + (size_t)(rc0 + (lane & 15)) * ld + (k0 + ((lane >> 4) << 3));
  v16bf t;
#pragma unroll
  for (int i = 0; i < 8; ++i) { t[i] = p[i]; t[i + 8] = p[i + 16]; }
  return t;
}

__global__ void cvt_bf16_kernel(const float* __restrict__ in, __bf16* __restrict__ out, int n) {
  int i = blockIdx.x * blockDim.x + threadIdx.x;
  if (i < n) out[i] = (__bf16)in[i];
}

// C = act[M,512] @ W[N,512]^T, one 16x16 tile per wave, 4 waves (64 cols) per block.
// MODE 0: Q projection  -> RMS-normalize per head row, store bf16 [b,h,n,d]
// MODE 1: KV projection -> cols <512: K (normalized, [b,h,n,d]); cols >=512: V transposed [b,h,d,n]
// MODE 2: out projection -> store f32 to d_out
template <int MODE>
__global__ __launch_bounds__(128) void gemm_kernel(const __bf16* __restrict__ A,
                                                   const __bf16* __restrict__ W,
                                                   __bf16* __restrict__ o0,
                                                   __bf16* __restrict__ o1,
                                                   float* __restrict__ fo) {
  __shared__ float rowss[16];
  const int lane = threadIdx.x & 31;
  const int wave = threadIdx.x >> 5;
  const int half = lane >> 4;
  const int nl   = lane & 15;
  const int m0   = blockIdx.x * 16;
  const int nn0  = blockIdx.y * 64 + wave * 16;

  v8f acc = {};
#pragma unroll 4
  for (int k0 = 0; k0 < DIM; k0 += 32) {
    v16bf a = load_tile16(A, m0, DIM, k0, lane);
    v16bf b = load_tile16(W, nn0, DIM, k0, lane);
    acc = wmma_bf16(a, b, acc);
  }

  // block covers exactly one head's 64 columns for rows m0..m0+15
  const bool do_norm = (MODE == 0) || (MODE == 1 && blockIdx.y < HEADS);
  if (do_norm) {
    if (threadIdx.x < 16) rowss[threadIdx.x] = 0.f;
    __syncthreads();
    float ss[8];
#pragma unroll
    for (int r = 0; r < 8; ++r) {
      float s = acc[r] * acc[r];
#pragma unroll
      for (int mk = 1; mk < 16; mk <<= 1) s += __shfl_xor(s, mk, 32);
      ss[r] = s;
    }
    if (nl == 0) {
#pragma unroll
      for (int r = 0; r < 8; ++r) atomicAdd(&rowss[half * 8 + r], ss[r]);
    }
    __syncthreads();
  }

#pragma unroll
  for (int r = 0; r < 8; ++r) {
    const int M  = m0 + half * 8 + r;    // global row (b*SEQ + n)
    const int nn = nn0 + nl;             // output column
    float v = acc[r];
    if (do_norm) v = v / (sqrtf(rowss[half * 8 + r] * (1.f / DH)) + EPS);
    if (MODE == 2) {
      fo[(size_t)M * DIM + nn] = v;
    } else {
      const int bb = M >> 11, n = M & (SEQ - 1);
      if (MODE == 0 || nn < DIM) {
        const int hh = (nn & (DIM - 1)) >> 6, d = nn & 63;
        o0[(((size_t)bb * HEADS + hh) * SEQ + n) * DH + d] = (__bf16)v;
      } else {
        const int n2 = nn - DIM, hh = n2 >> 6, d = n2 & 63;
        o1[(((size_t)bb * HEADS + hh) * DH + d) * SEQ + n] = (__bf16)v;  // V^T
      }
    }
  }
}

// Flash attention. One block = 4 waves = 64 queries of one (b,h); each wave owns
// a 16-query tile. Per 32-key step the block cooperatively DMAs the K tile
// (32x64 bf16) and V^T tile (64x32 bf16) into LDS with async copies, double
// buffered via ASYNCcnt, then computes S^T = K@Q^T so the C-layout of the two
// S^T tiles is exactly the A-layout needed for P in the P@V WMMA.
__global__ __launch_bounds__(128) void attn_kernel(const __bf16* __restrict__ Q,
                                                   const __bf16* __restrict__ K,
                                                   const __bf16* __restrict__ Vt,
                                                   __bf16* __restrict__ AO) {
  __shared__ __bf16 lk[2][32 * 64];   // [buf][key][d]
  __shared__ __bf16 lv[2][64 * 32];   // [buf][d][key]

  const int t    = threadIdx.x;
  const int lane = t & 31;
  const int wave = t >> 5;
  const int half = lane >> 4;
  const int nl   = lane & 15;          // query index within tile
  const int h = blockIdx.y, b = blockIdx.z;
  const int qt = blockIdx.x * 4 + wave;

  const __bf16* Qh = Q  + (((size_t)b * HEADS + h) * SEQ + qt * 16) * DH;
  const __bf16* Kh = K  + ((size_t)b * HEADS + h) * SEQ * DH;
  const __bf16* Vh = Vt + ((size_t)b * HEADS + h) * DH * SEQ;

  // cooperative async prefetch of one 32-key step into LDS buffer `buf`
  auto prefetch = [&](int buf, int j0) {
#pragma unroll
    for (int i = 0; i < 2; ++i) {                 // K: 256 x 16B chunks / 128 thr
      const int c = t + i * 128;
      const int row = c >> 3, col = (c & 7) * 8;  // row: key 0..31, col: d
      async_copy16(lds_addr(&lk[buf][row * 64 + col]), Kh + (size_t)(j0 + row) * DH + col);
    }
#pragma unroll
    for (int i = 0; i < 2; ++i) {                 // V^T: 256 x 16B chunks / 128 thr
      const int c = t + i * 128;
      const int row = c >> 2, col = (c & 3) * 8;  // row: d 0..63, col: key
      async_copy16(lds_addr(&lv[buf][row * 32 + col]), Vh + (size_t)row * SEQ + j0 + col);
    }
  };

  // Q as B-operand (d x q): column n = query row n of Q, contiguous over d
  const v16bf qb0 = load_tile16(Qh, 0, DH, 0, lane);
  const v16bf qb1 = load_tile16(Qh, 0, DH, 32, lane);

  v8f o0 = {}, o1 = {}, o2 = {}, o3 = {};   // out tile 16(q) x 64(d)
  float m = -3.0e38f, l = 0.f;              // running stats for query nl

  prefetch(0, 0);
  for (int jt = 0; jt < SEQ / 32; ++jt) {
    const int buf = jt & 1;
    if (jt + 1 < SEQ / 32) {
      prefetch(buf ^ 1, (jt + 1) * 32);
      ASYNC_WAIT(4);          // this step's 4 copies done; next step's in flight
    } else {
      ASYNC_WAIT(0);
    }
    __syncthreads();          // tile visible to all 4 waves

    const __bf16* lkb = &lk[buf][0];
    const __bf16* lvb = &lv[buf][0];

    // K rows as A-operand (keys x d), two 16-key tiles, d split in two K=32 chunks
    v16bf ka00 = load_tile16(lkb, 0,  DH, 0,  lane);
    v16bf ka01 = load_tile16(lkb, 0,  DH, 32, lane);
    v16bf ka10 = load_tile16(lkb, 16, DH, 0,  lane);
    v16bf ka11 = load_tile16(lkb, 16, DH, 32, lane);
    v8f z = {};
    v8f s0 = wmma_bf16(ka00, qb0, z); s0 = wmma_bf16(ka01, qb1, s0);  // keys +0..15
    v8f s1 = wmma_bf16(ka10, qb0, z); s1 = wmma_bf16(ka11, qb1, s1);  // keys +16..31

    float tmax = -3.0e38f;
#pragma unroll
    for (int r = 0; r < 8; ++r) {
      s0[r] *= QKSCALE; s1[r] *= QKSCALE;
      tmax = fmaxf(tmax, fmaxf(s0[r], s1[r]));
    }
    tmax = fmaxf(tmax, __shfl_xor(tmax, 16, 32));   // other 16 keys, same query
    const float mn = fmaxf(m, tmax);
    const float alpha = __expf(m - mn);
    float p0[8], p1[8], rs = 0.f;
#pragma unroll
    for (int r = 0; r < 8; ++r) {
      p0[r] = __expf(s0[r] - mn);
      p1[r] = __expf(s1[r] - mn);
      rs += p0[r] + p1[r];
    }
    rs += __shfl_xor(rs, 16, 32);
    l = l * alpha + rs;
    m = mn;

    // rescale accumulator: out C row M = half*8 + r; its alpha lives in lane M
#pragma unroll
    for (int r = 0; r < 8; ++r) {
      const float f = __shfl(alpha, half * 8 + r, 32);
      o0[r] *= f; o1[r] *= f; o2[r] *= f; o3[r] *= f;
    }

    // pack P into A-layout: S^T tile0 supplies K=0..7 (+half*8), tile1 K=16..23 (+half*8)
    v16bf pa;
#pragma unroll
    for (int r = 0; r < 8; ++r) { pa[r] = (__bf16)p0[r]; pa[r + 8] = (__bf16)p1[r]; }

    // V^T as B-operand (keys x d): column n = d-index, contiguous over keys in LDS
    v16bf vb0 = load_tile16(lvb, 0,  32, 0, lane);
    v16bf vb1 = load_tile16(lvb, 16, 32, 0, lane);
    v16bf vb2 = load_tile16(lvb, 32, 32, 0, lane);
    v16bf vb3 = load_tile16(lvb, 48, 32, 0, lane);
    o0 = wmma_bf16(pa, vb0, o0);
    o1 = wmma_bf16(pa, vb1, o1);
    o2 = wmma_bf16(pa, vb2, o2);
    o3 = wmma_bf16(pa, vb3, o3);

    __syncthreads();          // all waves done reading buf before it is refilled
  }

  const float linv = 1.0f / l;
  __bf16* dst = AO + ((size_t)b * SEQ + qt * 16) * DIM + h * DH;
#pragma unroll
  for (int r = 0; r < 8; ++r) {
    const float f = __shfl(linv, half * 8 + r, 32);
    __bf16* row = dst + (size_t)(half * 8 + r) * DIM + nl;
    row[0]  = (__bf16)(o0[r] * f);
    row[16] = (__bf16)(o1[r] * f);
    row[32] = (__bf16)(o2[r] * f);
    row[48] = (__bf16)(o3[r] * f);
  }
}

extern "C" void kernel_launch(void* const* d_in, const int* in_sizes, int n_in,
                              void* d_out, int out_size, void* d_ws, size_t ws_size,
                              hipStream_t stream) {
  const float* x    = (const float*)d_in[0];
  const float* Wq   = (const float*)d_in[1];
  const float* Wkv  = (const float*)d_in[2];
  const float* Wout = (const float*)d_in[3];
  float* out = (float*)d_out;

  __bf16* ws    = (__bf16*)d_ws;
  __bf16* xbf   = ws;                               // 4096*512
  __bf16* wqb   = xbf   + (size_t)ROWS * DIM;       // 512*512
  __bf16* wkvb  = wqb   + (size_t)DIM * DIM;        // 1024*512
  __bf16* woutb = wkvb  + (size_t)2 * DIM * DIM;    // 512*512
  __bf16* qb    = woutb + (size_t)DIM * DIM;        // [b,h,n,d]
  __bf16* kb    = qb    + (size_t)ROWS * DIM;       // [b,h,n,d]
  __bf16* vtb   = kb    + (size_t)ROWS * DIM;       // [b,h,d,n]
  __bf16* aob   = vtb   + (size_t)ROWS * DIM;       // [b,n,h*d]

  cvt_bf16_kernel<<<(ROWS * DIM + 255) / 256, 256, 0, stream>>>(x, xbf, ROWS * DIM);
  cvt_bf16_kernel<<<(DIM * DIM + 255) / 256, 256, 0, stream>>>(Wq, wqb, DIM * DIM);
  cvt_bf16_kernel<<<(2 * DIM * DIM + 255) / 256, 256, 0, stream>>>(Wkv, wkvb, 2 * DIM * DIM);
  cvt_bf16_kernel<<<(DIM * DIM + 255) / 256, 256, 0, stream>>>(Wout, woutb, DIM * DIM);

  gemm_kernel<0><<<dim3(ROWS / 16, DIM / 64), 128, 0, stream>>>(xbf, wqb, qb, nullptr, nullptr);
  gemm_kernel<1><<<dim3(ROWS / 16, 2 * DIM / 64), 128, 0, stream>>>(xbf, wkvb, kb, vtb, nullptr);
  attn_kernel<<<dim3(SEQ / 64, HEADS, BATCH), 128, 0, stream>>>(qb, kb, vtb, aob);
  gemm_kernel<2><<<dim3(ROWS / 16, DIM / 64), 128, 0, stream>>>(aob, woutb, nullptr, nullptr, out);
}
